// FlaxAttention_18872086299216
// MI455X (gfx1250) — matrix operand order repaired and checked by reading, not compiled, not run on hardware
//
#include <hip/hip_runtime.h>
#include <hip/hip_bf16.h>

#define B_     4
#define S_     2048
#define QD     512          // query dim == inner dim
#define HEADS_ 8
#define DH     64
#define ROWS   (B_ * S_)    // 8192

typedef __bf16 bf16;
typedef __attribute__((ext_vector_type(16))) __bf16 v16bf;
typedef __attribute__((ext_vector_type(8)))  __bf16 v8bf;
typedef __attribute__((ext_vector_type(8)))  float  v8f;

__device__ __forceinline__ v8f wmma_bf16(v16bf a, v16bf b, v8f c) {
    return __builtin_amdgcn_wmma_f32_16x16x32_bf16(
        false, a, false, b, (short)0, c, false, false);
}

// ---- CDNA5 async global->LDS copy (ASYNCcnt-tracked) ----------------------
__device__ __forceinline__ unsigned lds_off(const void* p) {
    return (unsigned)(size_t)p;          // flat LDS addr: low 32 bits = LDS offset
}
__device__ __forceinline__ void async_b128(unsigned dst_lds, const void* src_global) {
    asm volatile("global_load_async_to_lds_b128 %0, %1, off"
                 :: "v"(dst_lds), "v"((unsigned long long)(size_t)src_global)
                 : "memory");
}
#define S_WAIT_ASYNC(n) asm volatile("s_wait_asynccnt " n ::: "memory")

// ---- fragment packing (ISA 7.12.2, wave32) --------------------------------
// A: 16x32 (MxK) bf16 row-major. lane: row = lane&15, half = lane>>4.
__device__ __forceinline__ v16bf pack_a_bf16(const bf16* __restrict__ src, int ld, int lane) {
    const int row = lane & 15, half = lane >> 4;
    const bf16* p = src + (size_t)row * ld + 8 * half;
    v16bf a;
#pragma unroll
    for (int i = 0; i < 8; ++i)  a[i] = p[i];
#pragma unroll
    for (int i = 8; i < 16; ++i) a[i] = p[i + 8];
    return a;
}
// B: 32x16 (KxN), N-major source: B[k][n] = src[n*ld + k]. Two 16B runs/lane (k 8h..8h+7, 16+8h..).
__device__ __forceinline__ v16bf pack_bT_bf16(const bf16* __restrict__ src, int ld, int lane) {
    const int n = lane & 15, half = lane >> 4;
    const bf16* p = src + (size_t)n * ld + 8 * half;
    v16bf b;
#pragma unroll
    for (int i = 0; i < 8; ++i)  b[i] = p[i];
#pragma unroll
    for (int i = 8; i < 16; ++i) b[i] = p[i + 8];
    return b;
}

// ---- one-time precision/layout conversion ---------------------------------
__global__ __launch_bounds__(256) void cvt_hs(const float* __restrict__ in, bf16* __restrict__ out) {
    const size_t i = ((size_t)blockIdx.x * blockDim.x + threadIdx.x) * 8;
    const float4 v0 = *(const float4*)(in + i);
    const float4 v1 = *(const float4*)(in + i + 4);
    v8bf o;
    o[0] = (__bf16)v0.x; o[1] = (__bf16)v0.y; o[2] = (__bf16)v0.z; o[3] = (__bf16)v0.w;
    o[4] = (__bf16)v1.x; o[5] = (__bf16)v1.y; o[6] = (__bf16)v1.z; o[7] = (__bf16)v1.w;
    *(v8bf*)(out + i) = o;
}
__global__ __launch_bounds__(256) void cvt_wT(const float* __restrict__ W, bf16* __restrict__ Wt) {
    const int idx = blockIdx.x * blockDim.x + threadIdx.x;   // 512*512
    const int k = idx >> 9, n = idx & 511;
    Wt[(size_t)n * QD + k] = (__bf16)W[idx];
}

// ---- GEMM: A(bf16)[M,512] @ Wt(bf16,N-major) -> bf16 ----------------------
// Block = 8 waves sharing one async-staged 16-row A tile in LDS; each wave
// computes 16x64 (tg = wave id). B fragments are register double-buffered.
__global__ __launch_bounds__(256)
void gemm16x64(const bf16* __restrict__ A, const bf16* __restrict__ Wt,
               bf16* __restrict__ out, float scale, int storeT) {
    __shared__ bf16 ldsA[16][QD];                 // 16 KB
    const int tid = threadIdx.x, lane = tid & 31;
    const int tm = blockIdx.x;                    // 512 row tiles
    const int tg = tid >> 5;                      // 8 col groups of 64

    const bf16* arow = A + (size_t)tm * 16 * QD;
    const unsigned la = lds_off(&ldsA[0][0]);
#pragma unroll
    for (int j = 0; j < 4; ++j) {                 // 1024 x 16B segments / 256 thr
        const int s = tid + j * 256;
        const int row = s >> 6, ob = (s & 63) * 16;
        async_b128(la + row * 1024 + ob, (const char*)(arow + (size_t)row * QD) + ob);
    }
    S_WAIT_ASYNC("0x0");
    __syncthreads();

    const bf16* aL = &ldsA[0][0];
    const bf16* w0 = Wt + (size_t)(tg * 64) * QD;
    v8f acc[4] = {{}, {}, {}, {}};
    v16bf a_cur = pack_a_bf16(aL, QD, lane);
    v16bf b_cur[4];
#pragma unroll
    for (int t = 0; t < 4; ++t) b_cur[t] = pack_bT_bf16(w0 + (size_t)(t * 16) * QD, QD, lane);

#pragma unroll 2
    for (int kk = 0; kk < QD; kk += 32) {
        v16bf a_nxt = {};
        v16bf b_nxt[4] = {};
        if (kk + 32 < QD) {                       // prefetch next k-step into regs
            a_nxt = pack_a_bf16(aL + kk + 32, QD, lane);
#pragma unroll
            for (int t = 0; t < 4; ++t)
                b_nxt[t] = pack_bT_bf16(w0 + (size_t)(t * 16) * QD + kk + 32, QD, lane);
        }
#pragma unroll
        for (int t = 0; t < 4; ++t) acc[t] = wmma_bf16(a_cur, b_cur[t], acc[t]);
        a_cur = a_nxt;
#pragma unroll
        for (int t = 0; t < 4; ++t) b_cur[t] = b_nxt[t];
    }

    const int col = lane & 15, half = lane >> 4;
    if (!storeT) {
        bf16* o = out + ((size_t)tm * 16 + 8 * half) * QD + tg * 64 + col;
#pragma unroll
        for (int t = 0; t < 4; ++t)
#pragma unroll
            for (int r = 0; r < 8; ++r)
                o[(size_t)r * QD + t * 16] = (__bf16)(acc[t][r] * scale);
    } else {                                      // V: write [B][QD][S]
        const int rowg = tm * 16, bb = rowg >> 11, sl = rowg & (S_ - 1);
#pragma unroll
        for (int t = 0; t < 4; ++t) {
            const int c = tg * 64 + t * 16 + col;
            v8bf o8;
#pragma unroll
            for (int r = 0; r < 8; ++r) o8[r] = (__bf16)(acc[t][r] * scale);
            *(v8bf*)(out + ((size_t)bb * QD + c) * S_ + sl + 8 * half) = o8;
        }
    }
}

// ---- flash attention: block = 8 waves on one (b,h), 128 queries -----------
// K and V^T staged per 64-key chunk into double-buffered LDS via async DMA.
__device__ __forceinline__ void stage_kv(unsigned ldsK, unsigned ldsV,
                                         const bf16* kp, const bf16* vtp,
                                         int c, int tid) {
#pragma unroll
    for (int j = 0; j < 2; ++j) {                 // 512 x 16B segs K + V / 256 thr
        const int s = tid + j * 256;
        const int row = s >> 3;                   // 0..63
        const int ob = (s & 7) * 16;              // byte offset in 128B row
        async_b128(ldsK + row * 128 + ob,
                   (const char*)(kp + (size_t)(c * 64 + row) * QD) + ob);
        async_b128(ldsV + row * 128 + ob,
                   (const char*)(vtp + (size_t)row * S_ + c * 64) + ob);
    }
}

__global__ __launch_bounds__(256)
void flash_attn(const bf16* __restrict__ Q, const bf16* __restrict__ K,
                const bf16* __restrict__ Vt, bf16* __restrict__ O) {
    __shared__ bf16 ldsK[2][64][64];              // 16 KB
    __shared__ bf16 ldsV[2][64][64];              // 16 KB
    const int tid = threadIdx.x, lane = tid & 31, wv = tid >> 5;
    const int bh = blockIdx.x >> 4;               // 32 (b,h) pairs
    const int qt = (blockIdx.x & 15) * 8 + wv;    // 128 q-tiles per (b,h)
    const int h = bh & (HEADS_ - 1), b = bh >> 3;
    const size_t rowbase = (size_t)b * S_;
    const bf16* qp  = Q  + (rowbase + (size_t)qt * 16) * QD + h * DH;
    const bf16* kp  = K  + rowbase * QD + h * DH;
    const bf16* vtp = Vt + ((size_t)b * QD + h * DH) * S_;   // [d][s]

    // Q^T as B operand (loop-invariant, in VGPRs)
    const v16bf qb0 = pack_bT_bf16(qp,      QD, lane);
    const v16bf qb1 = pack_bT_bf16(qp + 32, QD, lane);

    const int halfL = lane >> 4, col = lane & 15;
    float m = -3.0e38f, l = 0.0f;                 // stats for q = lane&15
    v8f acc[4] = {{}, {}, {}, {}};

    stage_kv(lds_off(&ldsK[0][0][0]), lds_off(&ldsV[0][0][0]), kp, vtp, 0, tid);

    const int NC = S_ / 64;                       // 32 chunks
    for (int c = 0; c < NC; ++c) {
        if (c + 1 < NC) {
            stage_kv(lds_off(&ldsK[(c + 1) & 1][0][0]),
                     lds_off(&ldsV[(c + 1) & 1][0][0]), kp, vtp, c + 1, tid);
            S_WAIT_ASYNC("0x4");                  // chunk c done, c+1 in flight
        } else {
            S_WAIT_ASYNC("0x0");
        }
        __syncthreads();

        const bf16* kb = &ldsK[c & 1][0][0];      // [key][d], ld=64
        const bf16* vb = &ldsV[c & 1][0][0];      // [d][key], ld=64
        // S^T tiles: A = K rows (LDS), B = Q^T (regs)
        v8f t0 = {}, t1 = {}, t2 = {}, t3 = {};
        t0 = wmma_bf16(pack_a_bf16(kb,                64, lane), qb0, t0);
        t0 = wmma_bf16(pack_a_bf16(kb + 32,           64, lane), qb1, t0);
        t1 = wmma_bf16(pack_a_bf16(kb + 16 * 64,      64, lane), qb0, t1);
        t1 = wmma_bf16(pack_a_bf16(kb + 16 * 64 + 32, 64, lane), qb1, t1);
        t2 = wmma_bf16(pack_a_bf16(kb + 32 * 64,      64, lane), qb0, t2);
        t2 = wmma_bf16(pack_a_bf16(kb + 32 * 64 + 32, 64, lane), qb1, t2);
        t3 = wmma_bf16(pack_a_bf16(kb + 48 * 64,      64, lane), qb0, t3);
        t3 = wmma_bf16(pack_a_bf16(kb + 48 * 64 + 32, 64, lane), qb1, t3);

        // online softmax over 64 keys: 32 in-lane + xor-16 partner
        float v = -3.0e38f;
#pragma unroll
        for (int r = 0; r < 8; ++r) {
            v = fmaxf(v, fmaxf(t0[r], t1[r]));
            v = fmaxf(v, fmaxf(t2[r], t3[r]));
        }
        v = fmaxf(v, __shfl_xor(v, 16, 32));
        const float mn = fmaxf(m, v);
        const float corr = __expf(m - mn);
        m = mn;
        float rs = 0.0f;
#pragma unroll
        for (int r = 0; r < 8; ++r) {
            t0[r] = __expf(t0[r] - mn);
            t1[r] = __expf(t1[r] - mn);
            t2[r] = __expf(t2[r] - mn);
            t3[r] = __expf(t3[r] - mn);
            rs += (t0[r] + t1[r]) + (t2[r] + t3[r]);
        }
        rs += __shfl_xor(rs, 16, 32);
        l = l * corr + rs;

        // rescale acc: row q = r + 8*halfL gets corr from lane q (bpermute)
#pragma unroll
        for (int r = 0; r < 8; ++r) {
            const float cE = __shfl(corr, r + 8 * halfL, 32);
            acc[0][r] *= cE; acc[1][r] *= cE; acc[2][r] *= cE; acc[3][r] *= cE;
        }

        // S^T D-layout == P A-layout: in-lane pack, no transpose needed
        v16bf pa0, pa1;
#pragma unroll
        for (int i = 0; i < 8; ++i) {
            pa0[i] = (__bf16)t0[i]; pa0[i + 8] = (__bf16)t1[i];
            pa1[i] = (__bf16)t2[i]; pa1[i + 8] = (__bf16)t3[i];
        }

        // P·V from LDS V^T (two 16B runs per lane)
#pragma unroll
        for (int t = 0; t < 4; ++t) {
            acc[t] = wmma_bf16(pa0, pack_bT_bf16(vb + (size_t)(t * 16) * 64,      64, lane), acc[t]);
            acc[t] = wmma_bf16(pa1, pack_bT_bf16(vb + (size_t)(t * 16) * 64 + 32, 64, lane), acc[t]);
        }
        __syncthreads();                          // protect buffer for next stage
    }

    const float linv = 1.0f / l;
    bf16* op = O + (rowbase + (size_t)qt * 16) * QD + h * DH + col;
#pragma unroll
    for (int r = 0; r < 8; ++r) {
        const float lE = __shfl(linv, r + 8 * halfL, 32);
        bf16* o = op + (size_t)(r + 8 * halfL) * QD;
        o[0]  = (__bf16)(acc[0][r] * lE);
        o[16] = (__bf16)(acc[1][r] * lE);
        o[32] = (__bf16)(acc[2][r] * lE);
        o[48] = (__bf16)(acc[3][r] * lE);
    }
}

// ---- output projection: attn(bf16) @ WoT(bf16) + bo -> f32 ----------------
__global__ __launch_bounds__(256)
void gemm_out(const bf16* __restrict__ A, const bf16* __restrict__ Wt,
              const float* __restrict__ bias, float* __restrict__ out) {
    __shared__ bf16 ldsA[16][QD];
    const int tid = threadIdx.x, lane = tid & 31;
    const int tm = blockIdx.x;
    const int tg = tid >> 5;

    const bf16* arow = A + (size_t)tm * 16 * QD;
    const unsigned la = lds_off(&ldsA[0][0]);
#pragma unroll
    for (int j = 0; j < 4; ++j) {
        const int s = tid + j * 256;
        const int row = s >> 6, ob = (s & 63) * 16;
        async_b128(la + row * 1024 + ob, (const char*)(arow + (size_t)row * QD) + ob);
    }
    S_WAIT_ASYNC("0x0");
    __syncthreads();

    const bf16* aL = &ldsA[0][0];
    const bf16* w0 = Wt + (size_t)(tg * 64) * QD;
    v8f acc[4] = {{}, {}, {}, {}};
    v16bf a_cur = pack_a_bf16(aL, QD, lane);
    v16bf b_cur[4];
#pragma unroll
    for (int t = 0; t < 4; ++t) b_cur[t] = pack_bT_bf16(w0 + (size_t)(t * 16) * QD, QD, lane);

#pragma unroll 2
    for (int kk = 0; kk < QD; kk += 32) {
        v16bf a_nxt = {};
        v16bf b_nxt[4] = {};
        if (kk + 32 < QD) {
            a_nxt = pack_a_bf16(aL + kk + 32, QD, lane);
#pragma unroll
            for (int t = 0; t < 4; ++t)
                b_nxt[t] = pack_bT_bf16(w0 + (size_t)(t * 16) * QD + kk + 32, QD, lane);
        }
#pragma unroll
        for (int t = 0; t < 4; ++t) acc[t] = wmma_bf16(a_cur, b_cur[t], acc[t]);
        a_cur = a_nxt;
#pragma unroll
        for (int t = 0; t < 4; ++t) b_cur[t] = b_nxt[t];
    }

    const int col = lane & 15, half = lane >> 4;
#pragma unroll
    for (int t = 0; t < 4; ++t) {
        const int c = tg * 64 + t * 16 + col;
        const float bb = bias[c];
        float* o = out + ((size_t)tm * 16 + 8 * half) * QD + c;
#pragma unroll
        for (int r = 0; r < 8; ++r) o[(size_t)r * QD] = acc[t][r] + bb;
    }
}

extern "C" void kernel_launch(void* const* d_in, const int* in_sizes, int n_in,
                              void* d_out, int out_size, void* d_ws, size_t ws_size,
                              hipStream_t stream) {
    const float* hs = (const float*)d_in[0];
    const float* Wq = (const float*)d_in[1];
    const float* Wk = (const float*)d_in[2];
    const float* Wv = (const float*)d_in[3];
    const float* Wo = (const float*)d_in[4];
    const float* bo = (const float*)d_in[5];
    float* out = (float*)d_out;

    bf16* Hb  = (bf16*)d_ws;                      // [8192,512]
    bf16* WqT = Hb  + (size_t)ROWS * QD;          // [512,512] N-major
    bf16* WkT = WqT + (size_t)QD * QD;
    bf16* WvT = WkT + (size_t)QD * QD;
    bf16* WoT = WvT + (size_t)QD * QD;
    bf16* Qb  = WoT + (size_t)QD * QD;            // [8192,512]
    bf16* Kb  = Qb  + (size_t)ROWS * QD;          // [8192,512]
    bf16* Vt  = Kb  + (size_t)ROWS * QD;          // [B][512][2048]
    bf16* Ab  = Vt  + (size_t)ROWS * QD;          // [8192,512]

    const dim3 blk(256);
    cvt_hs<<<(ROWS * QD) / (256 * 8), blk, 0, stream>>>(hs, Hb);
    cvt_wT<<<(QD * QD) / 256, blk, 0, stream>>>(Wq, WqT);
    cvt_wT<<<(QD * QD) / 256, blk, 0, stream>>>(Wk, WkT);
    cvt_wT<<<(QD * QD) / 256, blk, 0, stream>>>(Wv, WvT);
    cvt_wT<<<(QD * QD) / 256, blk, 0, stream>>>(Wo, WoT);

    const int gemmBlocks = ROWS / 16;                     // 512
    gemm16x64<<<gemmBlocks, blk, 0, stream>>>(Hb, WqT, Qb, 0.125f, 0);
    gemm16x64<<<gemmBlocks, blk, 0, stream>>>(Hb, WkT, Kb, 1.0f, 0);
    gemm16x64<<<gemmBlocks, blk, 0, stream>>>(Hb, WvT, Vt, 1.0f, 1);

    const int attnBlocks = (B_ * HEADS_) * (S_ / 128);    // 512
    flash_attn<<<attnBlocks, blk, 0, stream>>>(Qb, Kb, Vt, Ab);

    gemm_out<<<gemmBlocks, blk, 0, stream>>>(Ab, WoT, bo, out);
}
